// ACE_ODE_RNN_79104707657769
// MI455X (gfx1250) — matrix operand order repaired and checked by reading, not compiled, not run on hardware
//
#include <hip/hip_runtime.h>
#include <hip/hip_bf16.h>

typedef __attribute__((ext_vector_type(16))) _Float16 v16h;
typedef __attribute__((ext_vector_type(8)))  float    v8f;

#define D_    64
#define WID_  128
#define DD_   4096
#define NB_   16
#define SEQ_  16
#define IN_   32
#define OUT_  8
#define NSUB_ 4
#define NWG_  8

// ---- weight fragment table (each fragment: 32 lanes x 16 halves = 512 halves = 1KB)
#define FRAG_HALF 512
#define FB_F0 0      // f_w0: 8 m-tiles x 3 k-steps (K padded 65->96)
#define FB_G0 24     // g_w0: 8 x 3
#define FB_F1 48     // f_w1: 8 x 4
#define FB_G1 80     // g_w1: 8 x 4
#define FB_F2 112    // f_w2: 4 x 4
#define FB_G2 128    // g_w2: 256 x 4
#define NFRAG_TOTAL 1152

static constexpr size_t FRAG_BYTES = (size_t)NFRAG_TOTAL * FRAG_HALF * sizeof(_Float16); // 1.125 MB
static constexpr size_t KA_BYTES   = (size_t)NWG_ * 6 * DD_ * NB_ * sizeof(float);       // 12 MB
static constexpr size_t AA_BYTES   = (size_t)NWG_ * DD_ * NB_ * sizeof(float);           // 2 MB

// Tsit5 tableau
__constant__ float d_AT[5][5] = {
  { 0.161f, 0.f, 0.f, 0.f, 0.f },
  { -0.008480655492356989f, 0.335480655492357f, 0.f, 0.f, 0.f },
  { 2.8971530571054935f, -6.359448489975075f, 4.3622954328695815f, 0.f, 0.f },
  { 5.325864828439257f, -11.748883564062828f, 7.4955393428898365f, -0.09249506636175525f, 0.f },
  { 5.86145544294642f, -12.92096931784711f, 8.159367898576159f, -0.071584973281401f, -0.028269050394068383f }
};
__constant__ float d_CT[6] = { 0.0f, 0.161f, 0.327f, 0.9f, 0.9800255409045097f, 1.0f };
__constant__ float d_BT[6] = { 0.09646076681806523f, 0.01f, 0.4798896504144996f,
                               1.379008574103742f, -3.290069515436081f, 2.324710524099774f };

__device__ __forceinline__ v8f wmma32f16(v16h a, v16h b, v8f c) {
  // emits v_wmma_f32_16x16x32_f16 (wave32, 16x16 f32 accumulator across 8 VGPRs)
  return __builtin_amdgcn_wmma_f32_16x16x32_f16(false, a, false, b, (short)0, c, false, false);
}

// ---------------------------------------------------------------------------
// Prep: convert weights to f16 and swizzle into per-lane WMMA A-fragment order
// A 16x32 f16 layout: lane%16 = M row; lane/16 = g; half e: v=e/2,p=e&1,
// k_local = 16*g + e (contiguous k order; B uses the same).  One wave per fragment.
// ---------------------------------------------------------------------------
__global__ void prep_frags(const float* __restrict__ fw0, const float* __restrict__ gw0,
                           const float* __restrict__ fw1, const float* __restrict__ gw1,
                           const float* __restrict__ fw2, const float* __restrict__ gw2,
                           _Float16* __restrict__ frags)
{
  int fi = blockIdx.x;
  int lane = threadIdx.x;
  const float* src; int K, mt, kt;
  if (fi < FB_G0)      { src = fw0; K = 65;  int r = fi - FB_F0; mt = r / 3; kt = r % 3; }
  else if (fi < FB_F1) { src = gw0; K = 65;  int r = fi - FB_G0; mt = r / 3; kt = r % 3; }
  else if (fi < FB_G1) { src = fw1; K = 128; int r = fi - FB_F1; mt = r / 4; kt = r % 4; }
  else if (fi < FB_F2) { src = gw1; K = 128; int r = fi - FB_G1; mt = r / 4; kt = r % 4; }
  else if (fi < FB_G2) { src = fw2; K = 128; int r = fi - FB_F2; mt = r / 4; kt = r % 4; }
  else                 { src = gw2; K = 128; int r = fi - FB_G2; mt = r / 4; kt = r % 4; }
  int m = mt * 16 + (lane & 15);
  int g = lane >> 4;
  v16h val;
#pragma unroll
  for (int e = 0; e < 16; ++e) {
    int v = e >> 1, p = e & 1;
    int kl = g * 16 + e;
    int k = kt * 32 + kl;
    val[e] = (_Float16)((k < K) ? src[m * K + k] : 0.0f);
  }
  *(v16h*)(frags + (size_t)fi * FRAG_HALF + (size_t)lane * 16) = val;
}

// ---------------------------------------------------------------------------
// Main persistent kernel: 8 workgroups x 256 threads (8 waves), 16 batch/WG.
// ---------------------------------------------------------------------------
__global__ __launch_bounds__(256) void ace_main(
  const float* __restrict__ ts, const float* __restrict__ obs,
  const _Float16* __restrict__ frags,
  const float* __restrict__ fb0, const float* __restrict__ fb1, const float* __restrict__ fb2,
  const float* __restrict__ gb0, const float* __restrict__ gb1, const float* __restrict__ gb2,
  const float* __restrict__ fscale, const float* __restrict__ gscale,
  const float* __restrict__ rwh, const float* __restrict__ rwx, const float* __restrict__ rb,
  const float* __restrict__ ow0, const float* __restrict__ ob0,
  const float* __restrict__ ow1, const float* __restrict__ ob1,
  const float* __restrict__ ow2, const float* __restrict__ ob2,
  float* __restrict__ ka_g, float* __restrict__ aa_g,
  float* __restrict__ out)
{
  __shared__ float hh[D_][NB_];
  __shared__ float hsb[D_][NB_];
  __shared__ float khs[6][D_][NB_];                                 // reused as o-MLP scratch
  __shared__ _Float16 bf0[3 * FRAG_HALF] __attribute__((aligned(32))); // layer0 B frags (K=96)
  __shared__ _Float16 z1f[4 * FRAG_HALF] __attribute__((aligned(32)));
  __shared__ _Float16 z1g[4 * FRAG_HALF] __attribute__((aligned(32)));
  __shared__ _Float16 z2f[4 * FRAG_HALF] __attribute__((aligned(32)));
  __shared__ _Float16 z2g[4 * FRAG_HALF] __attribute__((aligned(32)));
  __shared__ float xc[IN_][NB_];
  __shared__ float dtc[NB_], t0c[NB_], nrm[NB_];

  const int tid  = threadIdx.x;
  const int lane = tid & 31;
  const int wv   = tid >> 5;
  const int wg   = blockIdx.x;
  const int b0   = wg * NB_;
  float* ka = ka_g + (size_t)wg * 6 * DD_ * NB_;
  float* aa = aa_g + (size_t)wg * DD_ * NB_;
  const float sf = fscale[0], sg = gscale[0];

  // zero layer0 B buffer (pad rows k=65..95 must stay zero forever)
  for (int i = tid; i < 3 * FRAG_HALF; i += 256) bf0[i] = (_Float16)0.0f;

  // h0 = tanh(rnn_wx @ obs[:,0] + b)
  for (int r = 0; r < 4; ++r) {
    int d = (tid >> 4) + 16 * r, n = tid & 15;
    float acc = rb[d];
    for (int e = 0; e < IN_; ++e) acc += rwx[d * IN_ + e] * obs[((b0 + n) * SEQ_ + 0) * IN_ + e];
    hh[d][n] = tanhf(acc);
  }
  __syncthreads();
  if (tid < NB_) {
    float s = 0.f;
    for (int d2 = 0; d2 < D_; ++d2) { float v = hh[d2][tid]; s += v * v; }
    nrm[tid] = sqrtf(s) + 1e-8f;
  }
  __syncthreads();
  for (int it = 0; it < DD_ * NB_ / 256; ++it) {       // a0 = outer(hn,hn)
    int idx = tid + 256 * it;
    int n = idx & 15, m = idx >> 4;
    int i = m >> 6, j = m & 63;
    aa[idx] = hh[i][n] * hh[j][n] / (nrm[n] * nrm[n]);
  }
  __syncthreads();

  for (int st = 0; st < SEQ_ - 1; ++st) {
    if (tid < NB_) {
      float ta = ts[(b0 + tid) * SEQ_ + st], tb = ts[(b0 + tid) * SEQ_ + st + 1];
      dtc[tid] = (tb - ta) * 0.25f;
      t0c[tid] = ta;
    }
    for (int i = tid; i < IN_ * NB_; i += 256) {
      int e = i >> 4, n = i & 15;
      xc[e][n] = obs[((b0 + n) * SEQ_ + st + 1) * IN_ + e];
    }
    __syncthreads();

    for (int sub = 0; sub < NSUB_; ++sub) {
      for (int s = 0; s < 6; ++s) {
        // ---- staged h:  hs = hh + dt * sum_q AT[s-1][q] * kh[q]
        for (int r = 0; r < 4; ++r) {
          int d = (tid >> 4) + 16 * r, n = tid & 15;
          float v = hh[d][n];
          for (int q = 0; q < s; ++q) v += dtc[n] * d_AT[s - 1][q] * khs[q][d][n];
          hsb[d][n] = v;
        }
        __syncthreads();

        // ---- fused online-softmax row x hs dot -> hp -> B fragment (f16)
        for (int rr = 0; rr < 4; ++rr) {
          int i = (tid >> 4) + 16 * rr, n = tid & 15;
          float dt = dtc[n];
          float mx = -3.0e38f, den = 0.f, num = 0.f;
          int base = (i * 64) * NB_ + n;
          for (int j = 0; j < D_; ++j) {
            float v = aa[base + j * NB_];
            for (int q = 0; q < s; ++q)
              v += dt * d_AT[s - 1][q] * ka[(size_t)q * DD_ * NB_ + base + j * NB_];
            float hj = hsb[j][n];
            if (v > mx) { float c = __expf(mx - v); den *= c; num *= c; mx = v; }
            float e = __expf(v - mx);
            den += e; num += e * hj;
          }
          float hp = num / den;
          int kt = i >> 5, kl = i & 31, gg = kl >> 4, ee = kl & 15;
          bf0[(kt * 32 + gg * 16 + n) * 16 + ee] = (_Float16)hp;   // B layout: k=16g+e, lane=16g+n
        }
        if (tid < NB_) {                                            // time row k=64
          float tv = t0c[tid] + ((float)sub + d_CT[s]) * dtc[tid];
          bf0[(2 * 32 + tid) * 16 + 0] = (_Float16)tv;
        }
        __syncthreads();

        const int n = lane & 15, gsel = lane >> 4;

        // ===== layer 0: [128x96]@[96x16], f & g, wave wv owns M-tile wv =====
        {
          v16h bv[3];
#pragma unroll
          for (int kt = 0; kt < 3; ++kt)
            bv[kt] = *(const v16h*)&bf0[(kt * 32 + lane) * 16];
          v8f accf = {0.f,0.f,0.f,0.f,0.f,0.f,0.f,0.f};
          v8f accg = accf;
#pragma unroll
          for (int kt = 0; kt < 3; ++kt) {
            v16h af = *(const v16h*)(frags + (size_t)(FB_F0 + wv * 3 + kt) * FRAG_HALF + (size_t)lane * 16);
            accf = wmma32f16(af, bv[kt], accf);
            v16h ag = *(const v16h*)(frags + (size_t)(FB_G0 + wv * 3 + kt) * FRAG_HALF + (size_t)lane * 16);
            accg = wmma32f16(ag, bv[kt], accg);
          }
#pragma unroll
          for (int r = 0; r < 8; ++r) {
            int m = wv * 16 + r + 8 * gsel;
            int kt = m >> 5, kl = m & 31, gg = kl >> 4, ee = kl & 15;
            int off = (kt * 32 + gg * 16 + n) * 16 + ee;
            float vf = accf[r] + fb0[m]; vf = vf * (1.f / (1.f + __expf(-vf)));  // silu
            z1f[off] = (_Float16)vf;
            float vg = accg[r] + gb0[m]; vg = vg * (1.f / (1.f + __expf(-vg)));
            z1g[off] = (_Float16)vg;
          }
        }
        __syncthreads();

        // ===== layer 1: [128x128]@[128x16], f & g =====
        {
          v16h bvf[4], bvg[4];
#pragma unroll
          for (int kt = 0; kt < 4; ++kt) {
            bvf[kt] = *(const v16h*)&z1f[(kt * 32 + lane) * 16];
            bvg[kt] = *(const v16h*)&z1g[(kt * 32 + lane) * 16];
          }
          v8f accf = {0.f,0.f,0.f,0.f,0.f,0.f,0.f,0.f};
          v8f accg = accf;
#pragma unroll
          for (int kt = 0; kt < 4; ++kt) {
            v16h af = *(const v16h*)(frags + (size_t)(FB_F1 + wv * 4 + kt) * FRAG_HALF + (size_t)lane * 16);
            accf = wmma32f16(af, bvf[kt], accf);
            v16h ag = *(const v16h*)(frags + (size_t)(FB_G1 + wv * 4 + kt) * FRAG_HALF + (size_t)lane * 16);
            accg = wmma32f16(ag, bvg[kt], accg);
          }
#pragma unroll
          for (int r = 0; r < 8; ++r) {
            int m = wv * 16 + r + 8 * gsel;
            int kt = m >> 5, kl = m & 31, gg = kl >> 4, ee = kl & 15;
            int off = (kt * 32 + gg * 16 + n) * 16 + ee;
            float vf = accf[r] + fb1[m]; vf = vf * (1.f / (1.f + __expf(-vf)));
            z2f[off] = (_Float16)vf;
            float vg = accg[r] + gb1[m]; vg = vg * (1.f / (1.f + __expf(-vg)));
            z2g[off] = (_Float16)vg;
          }
        }
        __syncthreads();

        // ===== layer 2: f [64x128] -> kh (LDS);  g [4096x128] -> ka (global) =====
        {
          v16h bvf[4], bvg[4];
#pragma unroll
          for (int kt = 0; kt < 4; ++kt) {
            bvf[kt] = *(const v16h*)&z2f[(kt * 32 + lane) * 16];
            bvg[kt] = *(const v16h*)&z2g[(kt * 32 + lane) * 16];
          }
          if (wv < 4) {
            v8f acc = {0.f,0.f,0.f,0.f,0.f,0.f,0.f,0.f};
#pragma unroll
            for (int kt = 0; kt < 4; ++kt) {
              v16h af = *(const v16h*)(frags + (size_t)(FB_F2 + wv * 4 + kt) * FRAG_HALF + (size_t)lane * 16);
              acc = wmma32f16(af, bvf[kt], acc);
            }
#pragma unroll
            for (int r = 0; r < 8; ++r) {
              int m = wv * 16 + r + 8 * gsel;
              khs[s][m][n] = sf * tanhf(acc[r] + fb2[m]);
            }
          }
          float* kaS = ka + (size_t)s * DD_ * NB_;
          for (int tt = 0; tt < 32; ++tt) {               // 32 g-tiles per wave
            int mt = wv * 32 + tt;
            v8f acc = {0.f,0.f,0.f,0.f,0.f,0.f,0.f,0.f};
#pragma unroll
            for (int kt = 0; kt < 4; ++kt) {
              v16h ag = *(const v16h*)(frags + (size_t)(FB_G2 + mt * 4 + kt) * FRAG_HALF + (size_t)lane * 16);
              acc = wmma32f16(ag, bvg[kt], acc);
            }
#pragma unroll
            for (int r = 0; r < 8; ++r) {
              int m = mt * 16 + r + 8 * gsel;
              kaS[(size_t)m * NB_ + n] = sg * tanhf(acc[r] + gb2[m]);
            }
          }
        }
        __syncthreads();
      } // stages

      // ---- Tsit5 combine: h += dt*sum(BT*kh);  a += dt*sum(BT*ka)
      for (int r = 0; r < 4; ++r) {
        int d = (tid >> 4) + 16 * r, n = tid & 15;
        float v = hh[d][n];
        for (int q = 0; q < 6; ++q) v += dtc[n] * d_BT[q] * khs[q][d][n];
        hh[d][n] = v;
      }
      for (int it = 0; it < DD_ * NB_ / 256; ++it) {
        int idx = tid + 256 * it;
        int n = idx & 15;
        float v = aa[idx];
        for (int q = 0; q < 6; ++q) v += dtc[n] * d_BT[q] * ka[(size_t)q * DD_ * NB_ + idx];
        aa[idx] = v;
      }
      __syncthreads();
    } // sub

    // ---- RNN cell: h = tanh(Wh h' + Wx x + b); re-init a = outer(hn,hn)
    for (int r = 0; r < 4; ++r) {
      int d = (tid >> 4) + 16 * r, n = tid & 15;
      float acc = rb[d];
      for (int e = 0; e < D_; ++e)  acc += rwh[d * D_ + e] * hh[e][n];
      for (int e = 0; e < IN_; ++e) acc += rwx[d * IN_ + e] * xc[e][n];
      hsb[d][n] = tanhf(acc);
    }
    __syncthreads();
    if (tid < NB_) {
      float s2 = 0.f;
      for (int d2 = 0; d2 < D_; ++d2) { float v = hsb[d2][tid]; s2 += v * v; }
      nrm[tid] = sqrtf(s2) + 1e-8f;
    }
    __syncthreads();
    for (int r = 0; r < 4; ++r) {
      int d = (tid >> 4) + 16 * r, n = tid & 15;
      hh[d][n] = hsb[d][n];
    }
    for (int it = 0; it < DD_ * NB_ / 256; ++it) {
      int idx = tid + 256 * it;
      int n = idx & 15, m = idx >> 4;
      int i = m >> 6, j = m & 63;
      aa[idx] = hsb[i][n] * hsb[j][n] / (nrm[n] * nrm[n]);
    }
    __syncthreads();
  } // st

  // ---- output MLP (tanh hidden, linear out); reuse khs as scratch
  float* tA = &khs[0][0][0];
  float* tB = tA + WID_ * NB_;
  for (int r = 0; r < 8; ++r) {
    int m = (tid >> 4) + 16 * r, n = tid & 15;
    float acc = ob0[m];
    for (int e = 0; e < D_; ++e) acc += ow0[m * D_ + e] * hh[e][n];
    tA[m * NB_ + n] = tanhf(acc);
  }
  __syncthreads();
  for (int r = 0; r < 8; ++r) {
    int m = (tid >> 4) + 16 * r, n = tid & 15;
    float acc = ob1[m];
    for (int e = 0; e < WID_; ++e) acc += ow1[m * WID_ + e] * tA[e * NB_ + n];
    tB[m * NB_ + n] = tanhf(acc);
  }
  __syncthreads();
  if (tid < OUT_ * NB_) {
    int n = tid >> 3, o = tid & 7;
    float acc = ob2[o];
    for (int e = 0; e < WID_; ++e) acc += ow2[o * WID_ + e] * tB[e * NB_ + n];
    out[(b0 + n) * OUT_ + o] = acc;
  }
}

extern "C" void kernel_launch(void* const* d_in, const int* in_sizes, int n_in,
                              void* d_out, int out_size, void* d_ws, size_t ws_size,
                              hipStream_t stream) {
  (void)in_sizes; (void)n_in; (void)out_size; (void)ws_size;
  const float* ts  = (const float*)d_in[0];
  const float* obs = (const float*)d_in[1];
  const float* fw0 = (const float*)d_in[2];
  const float* fb0 = (const float*)d_in[3];
  const float* fw1 = (const float*)d_in[4];
  const float* fb1 = (const float*)d_in[5];
  const float* fw2 = (const float*)d_in[6];
  const float* fb2 = (const float*)d_in[7];
  const float* gw0 = (const float*)d_in[8];
  const float* gb0 = (const float*)d_in[9];
  const float* gw1 = (const float*)d_in[10];
  const float* gb1 = (const float*)d_in[11];
  const float* gw2 = (const float*)d_in[12];
  const float* gb2 = (const float*)d_in[13];
  const float* fsc = (const float*)d_in[14];
  const float* gsc = (const float*)d_in[15];
  const float* rwh = (const float*)d_in[16];
  const float* rwx = (const float*)d_in[17];
  const float* rb  = (const float*)d_in[18];
  const float* ow0 = (const float*)d_in[19];
  const float* ob0 = (const float*)d_in[20];
  const float* ow1 = (const float*)d_in[21];
  const float* ob1 = (const float*)d_in[22];
  const float* ow2 = (const float*)d_in[23];
  const float* ob2 = (const float*)d_in[24];
  float* out = (float*)d_out;

  char* base = (char*)d_ws;
  _Float16* frags = (_Float16*)base;
  float* ka = (float*)(base + FRAG_BYTES);                 // [8][6][4096][16] f32
  float* aa = (float*)(base + FRAG_BYTES + KA_BYTES);      // [8][4096][16]    f32

  prep_frags<<<NFRAG_TOTAL, 32, 0, stream>>>(fw0, gw0, fw1, gw1, fw2, gw2, frags);
  ace_main<<<NWG_, 256, 0, stream>>>(ts, obs, frags,
                                     fb0, fb1, fb2, gb0, gb1, gb2,
                                     fsc, gsc, rwh, rwx, rb,
                                     ow0, ob0, ow1, ob1, ow2, ob2,
                                     ka, aa, out);
}
